// _QuantizedLayerBase_86088324481052
// MI455X (gfx1250) — matrix operand order repaired and checked by reading, not compiled, not run on hardware
//
#include <hip/hip_runtime.h>
#include <stdint.h>

// ---------------------------------------------------------------------------
// Quantized linear: y[n,o] = scale[o] * sum_i x[n,i] * q[o,i]
//   x        : [8192, 4096] fp32
//   packed_w : [4096*4096/2] int32, one packed byte per element
//              (low nibble = even i, high nibble = odd i), value = nibble - 8
//   scale    : [4096] fp32
//   out      : [8192, 4096] fp32
//
// Fast path (ws_size >= 96MB):
//   prep_a: x -> f16, fragment-major A'  [512 tileM][128 tileK][32 lane][16]
//   prep_b: int4 -> f16 ints, fragment-major B' [256 tileO][128 tileK][32][16]
//   gemm  : pure b128 loads + v_wmma_f32_16x16x32_f16, 64x64 wave tile,
//           scale applied in f32 epilogue.
// Fallback (small ws): fused dequant GEMM (round-1 kernel).
// ---------------------------------------------------------------------------

#define N_TOK  8192
#define I_FEAT 4096
#define O_FEAT 4096
#define KT_CNT (I_FEAT / 32)   // 128 K-tiles
#define MT_CNT (N_TOK / 16)    // 512 M-tiles
#define OT_CNT (O_FEAT / 16)   // 256 O-tiles

typedef __attribute__((ext_vector_type(16))) _Float16 v16h;
typedef __attribute__((ext_vector_type(8)))  float    v8f;
typedef __attribute__((ext_vector_type(2)))  _Float16 h2;

// One packed byte -> two f16 values (nibble - 8), exact.
// (0x6400 | n) is f16(1024 + n); subtract 1032 -> n - 8.
__device__ __forceinline__ h2 nib_pair_to_h2(uint32_t w) {
    uint32_t bits = (w & 0x0000000Fu) | ((w & 0x000000F0u) << 12);
    bits |= 0x64006400u;
    h2 v = __builtin_bit_cast(h2, bits);
    h2 m;
    m[0] = (_Float16)1032.0f;
    m[1] = (_Float16)1032.0f;
    return v - m;
}

// B fragment (32x16 f16) in this code's contiguous k order:
//   lanes 0-15 : N = lane,    K = 0..15 ; lanes 16-31: N = lane-16, K = 16..31
__device__ __forceinline__ v16h load_b_frag(const uint32_t* __restrict__ wp,
                                            int col, int kbase) {
    const uint32_t* p = wp + (size_t)col * (I_FEAT / 2) + (kbase >> 1);
    uint4 q0 = *(const uint4*)(p);
    uint4 q1 = *(const uint4*)(p + 4);
    uint32_t ws[8] = {q0.x, q0.y, q0.z, q0.w, q1.x, q1.y, q1.z, q1.w};
    v16h r;
#pragma unroll
    for (int j = 0; j < 8; ++j) {
        h2 h = nib_pair_to_h2(ws[j]);
        r[2 * j]     = h[0];
        r[2 * j + 1] = h[1];
    }
    return r;
}

// A fragment (16x32 f16) in contiguous k order (k = 16*(lane/16) + s); xrow = &x[row][k0], khalf = 0 or 8.
__device__ __forceinline__ v16h load_a_frag(const float* __restrict__ xrow,
                                            int khalf) {
    float4 f0 = *(const float4*)(xrow + 2 * khalf);
    float4 f1 = *(const float4*)(xrow + 2 * khalf + 4);
    float4 f2 = *(const float4*)(xrow + 2 * khalf + 8);
    float4 f3 = *(const float4*)(xrow + 2 * khalf + 12);
    float fs[16] = {f0.x, f0.y, f0.z, f0.w, f1.x, f1.y, f1.z, f1.w,
                    f2.x, f2.y, f2.z, f2.w, f3.x, f3.y, f3.z, f3.w};
    v16h r;
#pragma unroll
    for (int j = 0; j < 16; ++j) r[j] = (_Float16)fs[j];
    return r;
}

// ---------------- pre-pass kernels (fragment-major swizzle) ----------------

// One wave per (tileM, tileK); writes 32B/lane coalesced.
__global__ __launch_bounds__(256) void prep_a(const float* __restrict__ x,
                                              v16h* __restrict__ ap) {
    const int lane = threadIdx.x & 31;
    const int wid  = blockIdx.x * 8 + (threadIdx.x >> 5);  // 512*128 waves
    const int tileM = wid / KT_CNT;
    const int tileK = wid % KT_CNT;
    const int row   = tileM * 16 + (lane & 15);
    const int khalf = (lane >> 4) * 8;
    const float* xr = x + (size_t)row * I_FEAT + tileK * 32;
    ap[(size_t)wid * 32 + lane] = load_a_frag(xr, khalf);
}

// One wave per (tileO, tileK); dequant once per weight.
__global__ __launch_bounds__(256) void prep_b(const uint32_t* __restrict__ wp,
                                              v16h* __restrict__ bp) {
    const int lane = threadIdx.x & 31;
    const int wid  = blockIdx.x * 8 + (threadIdx.x >> 5);  // 256*128 waves
    const int tileO = wid / KT_CNT;
    const int tileK = wid % KT_CNT;
    const int col   = tileO * 16 + (lane & 15);
    const int kbase = tileK * 32 + (lane >> 4) * 16;
    bp[(size_t)wid * 32 + lane] = load_b_frag(wp, col, kbase);
}

// ---------------- main GEMM on pre-swizzled operands -----------------------
// Block: 8 waves = 2(M) x 4(N).  Wave tile: 64x64 = 4x4 WMMA tiles.
// Block tile: 128 x 256.  Grid: (512/8, 256/16) = (64, 16).
__global__ __launch_bounds__(256) void gemm_swz(
    const v16h*  __restrict__ ap,
    const v16h*  __restrict__ bp,
    const float* __restrict__ scale,
    float*       __restrict__ out) {

    const int lane  = threadIdx.x & 31;
    const int wave  = threadIdx.x >> 5;
    const int waveM = wave >> 2;   // 0..1
    const int waveN = wave & 3;    // 0..3

    const int tileM0 = blockIdx.x * 8 + waveM * 4;   // 4 M tiles per wave
    const int tileO0 = blockIdx.y * 16 + waveN * 4;  // 4 O tiles per wave

    v8f acc[4][4];
#pragma unroll
    for (int mt = 0; mt < 4; ++mt)
#pragma unroll
        for (int nt = 0; nt < 4; ++nt) acc[mt][nt] = (v8f)(0.0f);

    // Loop-invariant scalar bases; per-lane offset folds into voffset.
    const v16h* abase[4];
    const v16h* bbase[4];
#pragma unroll
    for (int t = 0; t < 4; ++t) {
        abase[t] = ap + ((size_t)(tileM0 + t) * KT_CNT) * 32 + lane;
        bbase[t] = bp + ((size_t)(tileO0 + t) * KT_CNT) * 32 + lane;
    }

    for (int kt = 0; kt < KT_CNT; ++kt) {
        v16h bf[4];
#pragma unroll
        for (int nt = 0; nt < 4; ++nt) bf[nt] = bbase[nt][(size_t)kt * 32];

#pragma unroll
        for (int mt = 0; mt < 4; ++mt) {
            v16h a = abase[mt][(size_t)kt * 32];
#pragma unroll
            for (int nt = 0; nt < 4; ++nt) {
                acc[mt][nt] = __builtin_amdgcn_wmma_f32_16x16x32_f16(
                    false, a, false, bf[nt], (short)0, acc[mt][nt],
                    false, false);
            }
        }
    }

    // Epilogue: VGPR r holds M = 8*(lane>=16) + r, N = lane&15.
    const int crow = (lane >> 4) * 8;
    const int ccol = lane & 15;
#pragma unroll
    for (int nt = 0; nt < 4; ++nt) {
        const int o    = tileO0 * 16 + nt * 16 + ccol;
        const float sc = scale[o];
#pragma unroll
        for (int mt = 0; mt < 4; ++mt) {
#pragma unroll
            for (int r = 0; r < 8; ++r) {
                const int row = tileM0 * 16 + mt * 16 + crow + r;
                out[(size_t)row * O_FEAT + o] = acc[mt][nt][r] * sc;
            }
        }
    }
}

// ---------------- fallback: fused dequant GEMM (round-1) -------------------
__global__ __launch_bounds__(256) void quant_gemm_fused(
    const float*    __restrict__ x,
    const uint32_t* __restrict__ wp,
    const float*    __restrict__ scale,
    float*          __restrict__ out) {

    const int lane  = threadIdx.x & 31;
    const int wave  = threadIdx.x >> 5;
    const int waveM = wave >> 2;
    const int waveN = wave & 3;

    const int m0 = blockIdx.x * 64 + waveM * 32;
    const int n0 = blockIdx.y * 256 + waveN * 64;

    const int arow  = lane & 15;
    const int khalf = (lane >> 4) * 8;
    const int bcol  = lane & 15;
    const int bk    = (lane >> 4) * 16;

    v8f acc[2][4];
#pragma unroll
    for (int mt = 0; mt < 2; ++mt)
#pragma unroll
        for (int nt = 0; nt < 4; ++nt) acc[mt][nt] = (v8f)(0.0f);

    const float* xr0 = x + (size_t)(m0 + arow) * I_FEAT;
    const float* xr1 = x + (size_t)(m0 + 16 + arow) * I_FEAT;

    for (int k0 = 0; k0 < I_FEAT; k0 += 32) {
        v16h bf[4];
#pragma unroll
        for (int nt = 0; nt < 4; ++nt)
            bf[nt] = load_b_frag(wp, n0 + nt * 16 + bcol, k0 + bk);

        v16h a0 = load_a_frag(xr0 + k0, khalf);
        v16h a1 = load_a_frag(xr1 + k0, khalf);

#pragma unroll
        for (int nt = 0; nt < 4; ++nt) {
            acc[0][nt] = __builtin_amdgcn_wmma_f32_16x16x32_f16(
                false, a0, false, bf[nt], (short)0, acc[0][nt], false, false);
            acc[1][nt] = __builtin_amdgcn_wmma_f32_16x16x32_f16(
                false, a1, false, bf[nt], (short)0, acc[1][nt], false, false);
        }
    }

    const int crow = (lane >> 4) * 8;
    const int ccol = lane & 15;
#pragma unroll
    for (int nt = 0; nt < 4; ++nt) {
        const int o    = n0 + nt * 16 + ccol;
        const float sc = scale[o];
#pragma unroll
        for (int mt = 0; mt < 2; ++mt) {
#pragma unroll
            for (int r = 0; r < 8; ++r) {
                const int row = m0 + mt * 16 + crow + r;
                out[(size_t)row * O_FEAT + o] = acc[mt][nt][r] * sc;
            }
        }
    }
}

// ---------------------------------------------------------------------------
extern "C" void kernel_launch(void* const* d_in, const int* in_sizes, int n_in,
                              void* d_out, int out_size, void* d_ws, size_t ws_size,
                              hipStream_t stream) {
    const float*    xx = (const float*)d_in[0];
    const uint32_t* wp = (const uint32_t*)d_in[1];
    const float*    sc = (const float*)d_in[2];
    float*          yy = (float*)d_out;

    const size_t needA = (size_t)N_TOK * I_FEAT * sizeof(_Float16);   // 64 MB
    const size_t needB = (size_t)O_FEAT * I_FEAT * sizeof(_Float16);  // 32 MB

    if (ws_size >= needA + needB) {
        v16h* ap = (v16h*)d_ws;
        v16h* bpv = (v16h*)((char*)d_ws + needA);

        prep_a<<<dim3(MT_CNT * KT_CNT / 8), dim3(256), 0, stream>>>(xx, ap);
        prep_b<<<dim3(OT_CNT * KT_CNT / 8), dim3(256), 0, stream>>>(wp, bpv);
        gemm_swz<<<dim3(MT_CNT / 8, OT_CNT / 16), dim3(256), 0, stream>>>(
            ap, bpv, sc, yy);
    } else {
        quant_gemm_fused<<<dim3(N_TOK / 64, O_FEAT / 256), dim3(256), 0,
                           stream>>>(xx, wp, sc, yy);
    }
}
